// ScaledDotProductAttention_6700148982328
// MI455X (gfx1250) — matrix operand order repaired and checked by reading, not compiled, not run on hardware
//
#include <hip/hip_runtime.h>
#include <hip/hip_bf16.h>

// ---------------------------------------------------------------------------
// ScaledDotProductAttention for MI455X (gfx1250): bf16 WMMA GEMMs + fp32 softmax
// B=8, S=2048, D=1024.  d_out = [output fp32 (B*S*D)] ++ [attn fp32 (B*S*S)]
// Round 2: GLOBAL_LOAD_ASYNC_TO_LDS_B128 staging (ASYNCcnt) + double-buffered LDS
// ---------------------------------------------------------------------------

typedef __attribute__((ext_vector_type(16))) __bf16 v16bf;
typedef __attribute__((ext_vector_type(8)))  __bf16 v8bf;
typedef __attribute__((ext_vector_type(4)))  __bf16 v4bf;
typedef __attribute__((ext_vector_type(8)))  float  v8f;
typedef __attribute__((ext_vector_type(4)))  float  v4f;

#define LDP 72   // LDS row pitch in bf16 elements (64 data + 8 pad = 144B, 16B aligned)

union Frag16 { v16bf v; v8bf h[2]; };

// A fragment 16x32 bf16 (M x K) from row-major LDS tile.
// lanes 0-15: row M=lane,   VGPR0-3 = K[k0..k0+7],  VGPR4-7 = K[k0+16..k0+23]
// lanes16-31: row M=lane-16,VGPR0-3 = K[k0+8..+15], VGPR4-7 = K[k0+24..+31]
__device__ __forceinline__ v16bf lds_load_afrag(const __bf16* lds, int mBase, int k0, int lane) {
  int row = mBase + (lane & 15);
  int kb  = k0 + ((lane >> 4) << 3);
  const __bf16* p = lds + row * LDP + kb;
  Frag16 u;
  u.h[0] = *(const v8bf*)(p);
  u.h[1] = *(const v8bf*)(p + 16);
  return u.v;
}

// B fragment 32x16 bf16 (K x N) from LDS tile holding B^T row-major ([n][k]).
__device__ __forceinline__ v16bf lds_load_bfrag(const __bf16* lds, int nBase, int k0, int lane) {
  int row = nBase + (lane & 15);
  int kb  = k0 + ((lane >> 4) << 3);
  const __bf16* p = lds + row * LDP + kb;
  Frag16 u;
  u.h[0] = *(const v8bf*)(p);
  u.h[1] = *(const v8bf*)(p + 16);
  return u.v;
}

__device__ __forceinline__ v8f wmma_bf16(v16bf a, v16bf b, v8f c) {
  return __builtin_amdgcn_wmma_f32_16x16x32_bf16(false, a, false, b, (short)0, c, false, false);
}

// Stage a 128x64 fp32 tile (row-major) into LDS as bf16 (VGPR staging: needs cvt).
__device__ __forceinline__ void stage_f32_tile(const float* __restrict__ src, int pitch,
                                               int r0, int kk, __bf16* lds, int tid) {
  int cg = tid & 15;   // 16 groups of 4 floats per 64-wide row
  int rr = tid >> 4;   // 16 rows per pass
#pragma unroll
  for (int p = 0; p < 8; ++p) {
    int row = p * 16 + rr;
    v4f d = *(const v4f*)(src + (size_t)(r0 + row) * pitch + kk + cg * 4);
    v4bf o;
    o.x = (__bf16)d.x; o.y = (__bf16)d.y; o.z = (__bf16)d.z; o.w = (__bf16)d.w;
    *(v4bf*)(lds + row * LDP + cg * 4) = o;
  }
}

// Async-stage a 128x64 bf16 tile global->LDS, no VGPR data path (ASYNCcnt).
// GVS mode: SGPR 64-bit base + per-lane 32-bit byte offset. 1024 16B chunks,
// 4 per thread. LDS dest VGPR = low 32 bits of generic shared pointer.
__device__ __forceinline__ void async_stage_bf16_tile(const __bf16* src, int pitch,
                                                      int r0, int kk, __bf16* lds, int tid) {
  unsigned long long sbase = (unsigned long long)src;
#pragma unroll
  for (int p = 0; p < 4; ++p) {
    int c   = p * 256 + tid;
    int row = c >> 3, cg = c & 7;
    unsigned ldsOff = (unsigned)(unsigned long long)(const void*)(lds + row * LDP + cg * 8);
    unsigned voff   = (unsigned)(((size_t)(r0 + row) * pitch + kk + cg * 8) * 2u);
    asm volatile("global_load_async_to_lds_b128 %0, %1, %2"
                 :: "v"(ldsOff), "v"(voff), "s"(sbase) : "memory");
  }
}

__device__ __forceinline__ void wait_async() {
  asm volatile("s_wait_asynccnt 0x0" ::: "memory");
}

// One 64-wide K step: wave computes its 32x64 region = 2x4 WMMA tiles.
__device__ __forceinline__ void wave_mma_step(const __bf16* ldsA, const __bf16* ldsB,
                                              int wm, int wn, int lane, v8f acc[8]) {
#pragma unroll
  for (int ks = 0; ks < 64; ks += 32) {
    v16bf a0 = lds_load_afrag(ldsA, wm,      ks, lane);
    v16bf a1 = lds_load_afrag(ldsA, wm + 16, ks, lane);
    v16bf b0 = lds_load_bfrag(ldsB, wn,      ks, lane);
    v16bf b1 = lds_load_bfrag(ldsB, wn + 16, ks, lane);
    v16bf b2 = lds_load_bfrag(ldsB, wn + 32, ks, lane);
    v16bf b3 = lds_load_bfrag(ldsB, wn + 48, ks, lane);
    acc[0] = wmma_bf16(a0, b0, acc[0]);
    acc[1] = wmma_bf16(a0, b1, acc[1]);
    acc[2] = wmma_bf16(a0, b2, acc[2]);
    acc[3] = wmma_bf16(a0, b3, acc[3]);
    acc[4] = wmma_bf16(a1, b0, acc[4]);
    acc[5] = wmma_bf16(a1, b1, acc[5]);
    acc[6] = wmma_bf16(a1, b2, acc[6]);
    acc[7] = wmma_bf16(a1, b3, acc[7]);
  }
}

__device__ __forceinline__ void zero_acc(v8f acc[8]) {
#pragma unroll
  for (int i = 0; i < 8; ++i)
#pragma unroll
    for (int e = 0; e < 8; ++e) acc[i][e] = 0.0f;
}

// ---------------------------------------------------------------------------
// Kernel 1: projection  out[r,n] = sum_k X[r,k]*W[n,k] + bias[n]   (bf16 out)
// Double-buffered VGPR staging (fp32 source needs cvt).  grid = (8, 128)
// ---------------------------------------------------------------------------
__global__ __launch_bounds__(256) void proj_gemm_kernel(
    const float* __restrict__ X, const float* __restrict__ W,
    const float* __restrict__ bias, __bf16* __restrict__ out, int transposeV) {
  __shared__ alignas(16) __bf16 ldsA[2][128 * LDP];
  __shared__ alignas(16) __bf16 ldsB[2][128 * LDP];
  const int K = 1024, N = 1024;
  int n0 = blockIdx.x * 128;
  int r0 = blockIdx.y * 128;
  int tid = threadIdx.x, lane = tid & 31, wave = tid >> 5;
  int wm = (wave >> 1) * 32, wn = (wave & 1) * 64;
  v8f acc[8];
  zero_acc(acc);
  stage_f32_tile(X, K, r0, 0, ldsA[0], tid);
  stage_f32_tile(W, K, n0, 0, ldsB[0], tid);
  int cur = 0;
  for (int kk = 0; kk < K; kk += 64) {
    __syncthreads();                       // cur filled & visible
    if (kk + 64 < K) {                     // prefetch next tile into other buffer
      stage_f32_tile(X, K, r0, kk + 64, ldsA[cur ^ 1], tid);
      stage_f32_tile(W, K, n0, kk + 64, ldsB[cur ^ 1], tid);
    }
    wave_mma_step(ldsA[cur], ldsB[cur], wm, wn, lane, acc);
    __syncthreads();                       // all done reading cur
    cur ^= 1;
  }
  int cCol = lane & 15, cRow0 = (lane >> 4) * 8;
#pragma unroll
  for (int mi = 0; mi < 2; ++mi)
#pragma unroll
    for (int ni = 0; ni < 4; ++ni) {
      v8f a = acc[mi * 4 + ni];
      int gn = n0 + wn + ni * 16 + cCol;
      float bv = bias[gn];
      int rbase = r0 + wm + mi * 16 + cRow0;   // 8 consecutive rows
      if (!transposeV) {
#pragma unroll
        for (int j = 0; j < 8; ++j)
          out[(size_t)(rbase + j) * N + gn] = (__bf16)(a[j] + bv);
      } else {
        int bb = rbase >> 11, s = rbase & 2047;  // 128-row blocks never straddle batches
        v8bf o;
#pragma unroll
        for (int j = 0; j < 8; ++j) o[j] = (__bf16)(a[j] + bv);
        *(v8bf*)(out + (size_t)bb * 1024 * 2048 + (size_t)gn * 2048 + s) = o;
      }
    }
}

// ---------------------------------------------------------------------------
// Kernel 2: scores[b,s,t] = (qp[b,s,:] . kp[b,t,:]) / 32  (fp32), causal skip.
// Fully async double-buffered staging.  grid = (16 [t], 16 [s], 8)
// ---------------------------------------------------------------------------
__global__ __launch_bounds__(256) void score_gemm_kernel(
    const __bf16* __restrict__ qp, const __bf16* __restrict__ kp,
    float* __restrict__ scores) {
  if (blockIdx.x > blockIdx.y) return;   // whole-block uniform: EXEC stays full
  __shared__ alignas(16) __bf16 ldsA[2][128 * LDP];
  __shared__ alignas(16) __bf16 ldsB[2][128 * LDP];
  const int S = 2048, D = 1024;
  int t0 = blockIdx.x * 128, s0 = blockIdx.y * 128, b = blockIdx.z;
  const __bf16* A  = qp + (size_t)b * S * D;
  const __bf16* BT = kp + (size_t)b * S * D;
  int tid = threadIdx.x, lane = tid & 31, wave = tid >> 5;
  int wm = (wave >> 1) * 32, wn = (wave & 1) * 64;
  v8f acc[8];
  zero_acc(acc);
  async_stage_bf16_tile(A,  D, s0, 0, ldsA[0], tid);
  async_stage_bf16_tile(BT, D, t0, 0, ldsB[0], tid);
  int cur = 0;
  for (int kk = 0; kk < D; kk += 64) {
    wait_async();                          // this wave's fills for cur complete
    __syncthreads();                       // cross-wave visibility
    if (kk + 64 < D) {                     // async-prefetch next tile
      async_stage_bf16_tile(A,  D, s0, kk + 64, ldsA[cur ^ 1], tid);
      async_stage_bf16_tile(BT, D, t0, kk + 64, ldsB[cur ^ 1], tid);
    }
    wave_mma_step(ldsA[cur], ldsB[cur], wm, wn, lane, acc);
    __syncthreads();
    cur ^= 1;
  }
  float* sc = scores + (size_t)b * S * S;
  const float invT = 0.03125f;  // 1/32
  int cCol = lane & 15, cRow0 = (lane >> 4) * 8;
#pragma unroll
  for (int mi = 0; mi < 2; ++mi)
#pragma unroll
    for (int ni = 0; ni < 4; ++ni) {
      v8f a = acc[mi * 4 + ni];
      int gt = t0 + wn + ni * 16 + cCol;
      int gs = s0 + wm + mi * 16 + cRow0;
#pragma unroll
      for (int j = 0; j < 8; ++j)
        sc[(size_t)(gs + j) * S + gt] = a[j] * invT;
    }
}

// ---------------------------------------------------------------------------
// Kernel 3: in-place causal softmax over each row of attn [B*S rows, len S]
// masked (t>s) entries -> exactly 0 (matches exp(-1e16 - max) underflow)
// ---------------------------------------------------------------------------
__global__ __launch_bounds__(256) void softmax_kernel(float* __restrict__ attn) {
  const int S = 2048;
  int row = blockIdx.x;
  int b = row >> 11, s = row & 2047;
  float* p = attn + ((size_t)b * S + s) * S;
  int tid = threadIdx.x, lane = tid & 31, wave = tid >> 5;
  __shared__ float red[8];
  float x[8];
  float m = -3.0e38f;
#pragma unroll
  for (int i = 0; i < 8; ++i) {
    int idx = tid + i * 256;
    x[i] = (idx <= s) ? p[idx] : -3.0e38f;
    m = fmaxf(m, x[i]);
  }
#pragma unroll
  for (int off = 16; off > 0; off >>= 1) m = fmaxf(m, __shfl_xor(m, off));
  if (lane == 0) red[wave] = m;
  __syncthreads();
  m = red[0];
#pragma unroll
  for (int w = 1; w < 8; ++w) m = fmaxf(m, red[w]);
  __syncthreads();
  float sum = 0.0f;
#pragma unroll
  for (int i = 0; i < 8; ++i) {
    int idx = tid + i * 256;
    float e = (idx <= s) ? __expf(x[i] - m) : 0.0f;
    x[i] = e;
    sum += e;
  }
#pragma unroll
  for (int off = 16; off > 0; off >>= 1) sum += __shfl_xor(sum, off);
  if (lane == 0) red[wave] = sum;
  __syncthreads();
  float tot = 0.0f;
#pragma unroll
  for (int w = 0; w < 8; ++w) tot += red[w];
  float inv = 1.0f / tot;
#pragma unroll
  for (int i = 0; i < 8; ++i) {
    int idx = tid + i * 256;
    p[idx] = x[i] * inv;
  }
}

// ---------------------------------------------------------------------------
// Kernel 4: output[b,s,e] = sum_t attn[b,s,t] * vpT[b,e,t]   (fp32 out)
// A (attn fp32) VGPR-staged with cvt; B (vpT bf16) async-staged. Double-buffered.
// K capped at s0+128 (attn zero above diagonal).  grid = (8 [e], 16 [s], 8)
// ---------------------------------------------------------------------------
__global__ __launch_bounds__(256) void out_gemm_kernel(
    const float* __restrict__ attn, const __bf16* __restrict__ vpT,
    float* __restrict__ outp) {
  __shared__ alignas(16) __bf16 ldsA[2][128 * LDP];
  __shared__ alignas(16) __bf16 ldsB[2][128 * LDP];
  const int S = 2048, D = 1024;
  int n0 = blockIdx.x * 128;   // e
  int s0 = blockIdx.y * 128;   // s
  int b  = blockIdx.z;
  const float* A   = attn + (size_t)b * S * S;   // [S][S]
  const __bf16* BT = vpT  + (size_t)b * D * S;   // [D][S]
  int tid = threadIdx.x, lane = tid & 31, wave = tid >> 5;
  int wm = (wave >> 1) * 32, wn = (wave & 1) * 64;
  v8f acc[8];
  zero_acc(acc);
  int kend = s0 + 128;   // attn[s, t>s] == 0
  stage_f32_tile(A, S, s0, 0, ldsA[0], tid);
  async_stage_bf16_tile(BT, S, n0, 0, ldsB[0], tid);
  int cur = 0;
  for (int kk = 0; kk < kend; kk += 64) {
    wait_async();
    __syncthreads();
    if (kk + 64 < kend) {
      stage_f32_tile(A, S, s0, kk + 64, ldsA[cur ^ 1], tid);
      async_stage_bf16_tile(BT, S, n0, kk + 64, ldsB[cur ^ 1], tid);
    }
    wave_mma_step(ldsA[cur], ldsB[cur], wm, wn, lane, acc);
    __syncthreads();
    cur ^= 1;
  }
  int cCol = lane & 15, cRow0 = (lane >> 4) * 8;
#pragma unroll
  for (int mi = 0; mi < 2; ++mi)
#pragma unroll
    for (int ni = 0; ni < 4; ++ni) {
      v8f a = acc[mi * 4 + ni];
      int ge = n0 + wn + ni * 16 + cCol;
      int gs = s0 + wm + mi * 16 + cRow0;
#pragma unroll
      for (int j = 0; j < 8; ++j)
        outp[((size_t)b * S + gs + j) * D + ge] = a[j];
    }
}

// ---------------------------------------------------------------------------
extern "C" void kernel_launch(void* const* d_in, const int* in_sizes, int n_in,
                              void* d_out, int out_size, void* d_ws, size_t ws_size,
                              hipStream_t stream) {
  const float* q  = (const float*)d_in[0];
  const float* k  = (const float*)d_in[1];
  const float* v  = (const float*)d_in[2];
  const float* Wq = (const float*)d_in[3];
  const float* bq = (const float*)d_in[4];
  const float* Wk = (const float*)d_in[5];
  const float* bk = (const float*)d_in[6];
  const float* Wv = (const float*)d_in[7];
  const float* bv = (const float*)d_in[8];
  (void)d_in[9];  // boolean causal mask -- implicit in kernels
  (void)in_sizes; (void)n_in; (void)out_size; (void)ws_size;

  const size_t RD = (size_t)8 * 2048 * 1024;   // B*S*D = 16,777,216

  __bf16* qp  = (__bf16*)d_ws;       // 33.5 MB
  __bf16* kp  = qp + RD;             // 33.5 MB
  __bf16* vpT = kp + RD;             // 33.5 MB  ([B][D][S])

  float* outp = (float*)d_out;       // [B,S,D]
  float* attn = outp + RD;           // [B,S,S] scores -> softmax in place

  dim3 blk(256);

  dim3 gProj(1024 / 128, 16384 / 128);           // (8, 128)
  proj_gemm_kernel<<<gProj, blk, 0, stream>>>(q, Wq, bq, qp, 0);
  proj_gemm_kernel<<<gProj, blk, 0, stream>>>(k, Wk, bk, kp, 0);
  proj_gemm_kernel<<<gProj, blk, 0, stream>>>(v, Wv, bv, vpT, 1);

  dim3 gScore(2048 / 128, 2048 / 128, 8);        // (16, 16, 8)
  score_gemm_kernel<<<gScore, blk, 0, stream>>>(qp, kp, attn);

  softmax_kernel<<<dim3(8 * 2048), blk, 0, stream>>>(attn);

  dim3 gOut(1024 / 128, 2048 / 128, 8);          // (8, 16, 8)
  out_gemm_kernel<<<gOut, blk, 0, stream>>>(attn, vpT, outp);
}
